// DirectionalWeights_38732015075370
// MI455X (gfx1250) — compile-verified, matrix operand order checked by reading, not run on hardware
//
#include <hip/hip_runtime.h>
#include <hip/hip_bf16.h>

// Problem constants (match reference)
#define B_  2
#define N_  10000
#define E_  100000
#define F_  512
#define H_  512
#define BN_ (B_ * N_)           // 20000 node rows total
#define EPS_ 1e-5f

typedef __attribute__((ext_vector_type(16))) _Float16 v16h;
typedef __attribute__((ext_vector_type(8)))  _Float16 v8h;
typedef __attribute__((ext_vector_type(8)))  float    v8f;

// ---------------------------------------------------------------------------
// 0) zero the 4 segment accumulators (max/sum for both directions)
// ---------------------------------------------------------------------------
__global__ void zero_acc_kernel(float* acc, int n) {
    int i = blockIdx.x * blockDim.x + threadIdx.x;
    if (i < n) acc[i] = 0.0f;
}

// ---------------------------------------------------------------------------
// 1) convert W1,W2 (F x H, row-major f32) -> transposed f16 (H x F) so the
//    WMMA B-fragment (column n, contiguous K) is two b128 loads per lane.
// ---------------------------------------------------------------------------
__global__ void conv_w_kernel(const float* __restrict__ W1,
                              const float* __restrict__ W2,
                              _Float16* __restrict__ W1t,
                              _Float16* __restrict__ W2t) {
    int idx = blockIdx.x * blockDim.x + threadIdx.x;     // over 2*F*H
    if (idx >= 2 * F_ * H_) return;
    int mat = idx / (F_ * H_);
    int rem = idx - mat * (F_ * H_);                     // rem = h*F + f
    int h = rem / F_;
    int f = rem - h * F_;
    const float* W = mat ? W2 : W1;
    _Float16*  Wt = mat ? W2t : W1t;
    Wt[rem] = (_Float16)W[(size_t)f * H_ + h];
}

// ---------------------------------------------------------------------------
// 2) convert node features to f16
// ---------------------------------------------------------------------------
__global__ void conv_x_kernel(const float* __restrict__ X,
                              _Float16* __restrict__ Xh, size_t n) {
    size_t i = (size_t)blockIdx.x * blockDim.x + threadIdx.x;
    if (i < n) Xh[i] = (_Float16)X[i];
}

// ---------------------------------------------------------------------------
// 3) WMMA GEMM, 2x2 register-blocked: each wave32 computes a 32x32 output
//    region (four 16x16 tiles). Per k-step: 8 b128 loads -> 4 independent
//    v_wmma_f32_16x16x32_f16 (A-frags shared across the 2 column tiles,
//    B-frags shared across the 2 row tiles).
//    Fragment layouts per CDNA5 ISA 7.12.2:
//      A  (16x32 f16): lane = 16*khalf + m ; elems e[i]=A[m][8*khalf+i],
//                      e[8+i]=A[m][16+8*khalf+i]
//      B  (32x16 f16): mirror of A over columns (Wt is W^T, so row n of Wt)
//      C/D(16x16 f32): row = 8*(lane>>4) + r, col = lane&15
// ---------------------------------------------------------------------------
__global__ void gemm_wmma_kernel(const _Float16* __restrict__ Xh,
                                 const _Float16* __restrict__ W1t,
                                 const _Float16* __restrict__ W2t,
                                 const float* __restrict__ b1,
                                 const float* __restrict__ b2,
                                 float* __restrict__ Y /* 2 * BN * H */) {
    const int lane = threadIdx.x & 31;
    const int wave = threadIdx.x >> 5;
    const int task = blockIdx.x * 8 + wave;
    const int MT2 = BN_ / 32;                            // 625 row-blocks
    const int NT2 = H_ / 32;                             // 16  col-blocks
    const int TASKS_PER_MAT = MT2 * NT2;                 // 10000
    if (task >= 2 * TASKS_PER_MAT) return;

    const int mat = task / TASKS_PER_MAT;
    const int rem = task - mat * TASKS_PER_MAT;
    const int mt2 = rem / NT2;
    const int nt2 = rem - mt2 * NT2;

    const _Float16* __restrict__ Wt   = mat ? W2t : W1t;
    const float*    __restrict__ bias = mat ? b2  : b1;

    const int l15   = lane & 15;
    const int khalf = lane >> 4;
    const int m0 = mt2 * 32;                             // 32 output rows
    const int n0 = nt2 * 32;                             // 32 output cols

    const _Float16* __restrict__ arow0 = Xh + (size_t)(m0 + l15) * F_;
    const _Float16* __restrict__ arow1 = arow0 + (size_t)16 * F_;
    const _Float16* __restrict__ brow0 = Wt + (size_t)(n0 + l15) * F_;
    const _Float16* __restrict__ brow1 = brow0 + (size_t)16 * F_;
    const int koff = 8 * khalf;

    v8f acc00 = {}, acc01 = {}, acc10 = {}, acc11 = {};
#pragma unroll 2
    for (int k0 = 0; k0 < F_; k0 += 32) {
        __builtin_prefetch(arow0 + k0 + 64, 0, 0);
        __builtin_prefetch(arow1 + k0 + 64, 0, 0);

        v8h a0lo = *(const v8h*)(arow0 + k0 + koff);
        v8h a0hi = *(const v8h*)(arow0 + k0 + 16 + koff);
        v8h a1lo = *(const v8h*)(arow1 + k0 + koff);
        v8h a1hi = *(const v8h*)(arow1 + k0 + 16 + koff);
        v8h b0lo = *(const v8h*)(brow0 + k0 + koff);
        v8h b0hi = *(const v8h*)(brow0 + k0 + 16 + koff);
        v8h b1lo = *(const v8h*)(brow1 + k0 + koff);
        v8h b1hi = *(const v8h*)(brow1 + k0 + 16 + koff);

        v16h a0 = __builtin_shufflevector(a0lo, a0hi, 0,1,2,3,4,5,6,7,8,9,10,11,12,13,14,15);
        v16h a1 = __builtin_shufflevector(a1lo, a1hi, 0,1,2,3,4,5,6,7,8,9,10,11,12,13,14,15);
        v16h b0 = __builtin_shufflevector(b0lo, b0hi, 0,1,2,3,4,5,6,7,8,9,10,11,12,13,14,15);
        v16h b1 = __builtin_shufflevector(b1lo, b1hi, 0,1,2,3,4,5,6,7,8,9,10,11,12,13,14,15);

        acc00 = __builtin_amdgcn_wmma_f32_16x16x32_f16(false, a0, false, b0,
                                                       (short)0, acc00, false, false);
        acc01 = __builtin_amdgcn_wmma_f32_16x16x32_f16(false, a0, false, b1,
                                                       (short)0, acc01, false, false);
        acc10 = __builtin_amdgcn_wmma_f32_16x16x32_f16(false, a1, false, b0,
                                                       (short)0, acc10, false, false);
        acc11 = __builtin_amdgcn_wmma_f32_16x16x32_f16(false, a1, false, b1,
                                                       (short)0, acc11, false, false);
    }

    // epilogue: scatter the four C tiles + bias to fp32 Y
    float* __restrict__ out = Y + (size_t)mat * BN_ * H_;
    const int col0 = n0 + l15;          // tj = 0
    const int col1 = n0 + 16 + l15;     // tj = 1
    const float bv0 = bias[col0];
    const float bv1 = bias[col1];
    const int rb0 = m0 + 8 * khalf;     // ti = 0
    const int rb1 = m0 + 16 + 8 * khalf;// ti = 1
#pragma unroll
    for (int r = 0; r < 8; ++r) {
        out[(size_t)(rb0 + r) * H_ + col0] = acc00[r] + bv0;
        out[(size_t)(rb0 + r) * H_ + col1] = acc01[r] + bv1;
        out[(size_t)(rb1 + r) * H_ + col0] = acc10[r] + bv0;
        out[(size_t)(rb1 + r) * H_ + col1] = acc11[r] + bv1;
    }
}

// ---------------------------------------------------------------------------
// 4) in-place LayerNorm over H=512, one wave per row (rows 0..BN-1 -> mat0)
// ---------------------------------------------------------------------------
__global__ void ln_kernel(float* __restrict__ Y,
                          const float* __restrict__ g1, const float* __restrict__ bb1,
                          const float* __restrict__ g2, const float* __restrict__ bb2) {
    const int lane = threadIdx.x & 31;
    const int wave = threadIdx.x >> 5;
    const int row  = blockIdx.x * 8 + wave;              // 2*BN rows
    if (row >= 2 * BN_) return;

    const bool m2 = row >= BN_;
    const float* __restrict__ g  = m2 ? g2  : g1;
    const float* __restrict__ bb = m2 ? bb2 : bb1;
    float* __restrict__ x = Y + (size_t)row * H_;

    float s = 0.0f, sq = 0.0f;
    float v[16];
#pragma unroll
    for (int j = 0; j < 16; ++j) {
        v[j] = x[lane + 32 * j];
        s  += v[j];
        sq += v[j] * v[j];
    }
#pragma unroll
    for (int off = 16; off > 0; off >>= 1) {
        s  += __shfl_xor(s,  off, 32);
        sq += __shfl_xor(sq, off, 32);
    }
    const float mean = s * (1.0f / H_);
    const float var  = sq * (1.0f / H_) - mean * mean;
    const float inv  = rsqrtf(var + EPS_);
#pragma unroll
    for (int j = 0; j < 16; ++j) {
        const int col = lane + 32 * j;
        x[col] = (v[j] - mean) * inv * g[col] + bb[col];
    }
}

// ---------------------------------------------------------------------------
// 5) edge pass 1: one wave per edge. Dot relu(a1[s]+a2[d]) with W3 both ways,
//    form Vij/Vji, store raw values to d_out, atomic-max into segment maxima
//    (uint compare valid: relu output is non-negative).
// ---------------------------------------------------------------------------
__global__ void edge_pass1_kernel(const int* __restrict__ ei,
                                  const float* __restrict__ A,   // 2*BN*H
                                  const float* __restrict__ W3,
                                  const float* __restrict__ b3,
                                  const float* __restrict__ W4,
                                  const float* __restrict__ b4,
                                  float* __restrict__ out,       // 2*B*E
                                  float* __restrict__ maxIJ,
                                  float* __restrict__ maxJI) {
    const int lane = threadIdx.x & 31;
    const int wave = threadIdx.x >> 5;
    const int id   = blockIdx.x * 8 + wave;              // B*E edges
    if (id >= B_ * E_) return;

    const int b = id / E_;
    const int e = id - b * E_;
    const int src = ei[(size_t)(b * 2 + 0) * E_ + e];
    const int dst = ei[(size_t)(b * 2 + 1) * E_ + e];

    const float* __restrict__ A1 = A;
    const float* __restrict__ A2 = A + (size_t)BN_ * H_;
    const float* __restrict__ a1i = A1 + (size_t)(b * N_ + src) * H_;
    const float* __restrict__ a2j = A2 + (size_t)(b * N_ + dst) * H_;
    const float* __restrict__ a1j = A1 + (size_t)(b * N_ + dst) * H_;
    const float* __restrict__ a2i = A2 + (size_t)(b * N_ + src) * H_;

    float d1 = 0.0f, d2 = 0.0f;
#pragma unroll 4
    for (int j = 0; j < 16; ++j) {
        const int k = lane + 32 * j;
        const float w = W3[k];
        d1 += fmaxf(a1i[k] + a2j[k], 0.0f) * w;
        d2 += fmaxf(a1j[k] + a2i[k], 0.0f) * w;
    }
#pragma unroll
    for (int off = 16; off > 0; off >>= 1) {
        d1 += __shfl_xor(d1, off, 32);
        d2 += __shfl_xor(d2, off, 32);
    }
    if (lane == 0) {
        const float zij = d1 + b3[0];
        const float zji = d2 + b3[0];
        const float dd  = zij - zji;
        const float w4  = W4[0];
        const float vij = fmaxf( dd * w4 + b4[0], 0.0f);
        const float vji = fmaxf(-dd * w4 + b4[0], 0.0f);
        out[(size_t)b * E_ + e]                    = vij;
        out[(size_t)B_ * E_ + (size_t)b * E_ + e]  = vji;
        atomicMax((unsigned int*)&maxIJ[b * N_ + src], __float_as_uint(vij));
        atomicMax((unsigned int*)&maxJI[b * N_ + dst], __float_as_uint(vji));
    }
}

// ---------------------------------------------------------------------------
// 6) edge pass 2: exp(v - segmax), accumulate segment sums
// ---------------------------------------------------------------------------
__global__ void edge_pass2_kernel(const int* __restrict__ ei,
                                  float* __restrict__ out,
                                  const float* __restrict__ maxIJ,
                                  const float* __restrict__ maxJI,
                                  float* __restrict__ sumIJ,
                                  float* __restrict__ sumJI) {
    const int idx = blockIdx.x * blockDim.x + threadIdx.x;
    if (idx >= 2 * B_ * E_) return;
    const int half = idx / (B_ * E_);
    const int rem  = idx - half * (B_ * E_);
    const int b = rem / E_;
    const int e = rem - b * E_;
    const int seg = ei[(size_t)(b * 2 + half) * E_ + e];
    const float* mx = half ? maxJI : maxIJ;
    float*       sm = half ? sumJI : sumIJ;
    const float ex = expf(out[idx] - mx[b * N_ + seg]);
    out[idx] = ex;
    atomicAdd(&sm[b * N_ + seg], ex);
}

// ---------------------------------------------------------------------------
// 7) edge pass 3: normalize by segment sums
// ---------------------------------------------------------------------------
__global__ void edge_pass3_kernel(const int* __restrict__ ei,
                                  float* __restrict__ out,
                                  const float* __restrict__ sumIJ,
                                  const float* __restrict__ sumJI) {
    const int idx = blockIdx.x * blockDim.x + threadIdx.x;
    if (idx >= 2 * B_ * E_) return;
    const int half = idx / (B_ * E_);
    const int rem  = idx - half * (B_ * E_);
    const int b = rem / E_;
    const int e = rem - b * E_;
    const int seg = ei[(size_t)(b * 2 + half) * E_ + e];
    const float* sm = half ? sumJI : sumIJ;
    out[idx] = out[idx] / sm[b * N_ + seg];
}

// ---------------------------------------------------------------------------
extern "C" void kernel_launch(void* const* d_in, const int* in_sizes, int n_in,
                              void* d_out, int out_size, void* d_ws, size_t ws_size,
                              hipStream_t stream) {
    const float* X   = (const float*)d_in[0];
    const int*   ei  = (const int*)  d_in[1];
    // d_in[2] = num_nodes (compile-time constant here)
    const float* W1  = (const float*)d_in[3];
    const float* b1  = (const float*)d_in[4];
    const float* g1  = (const float*)d_in[5];
    const float* bb1 = (const float*)d_in[6];
    const float* W2  = (const float*)d_in[7];
    const float* b2  = (const float*)d_in[8];
    const float* g2  = (const float*)d_in[9];
    const float* bb2 = (const float*)d_in[10];
    const float* W3  = (const float*)d_in[11];
    const float* b3  = (const float*)d_in[12];
    const float* W4  = (const float*)d_in[13];
    const float* b4  = (const float*)d_in[14];
    float* out = (float*)d_out;

    // -------- workspace layout --------
    char* ws = (char*)d_ws;
    size_t off = 0;
    _Float16* Xh  = (_Float16*)(ws + off); off += (size_t)BN_ * F_ * sizeof(_Float16);
    _Float16* W1t = (_Float16*)(ws + off); off += (size_t)F_ * H_ * sizeof(_Float16);
    _Float16* W2t = (_Float16*)(ws + off); off += (size_t)F_ * H_ * sizeof(_Float16);
    float*    Y   = (float*)   (ws + off); off += (size_t)2 * BN_ * H_ * sizeof(float);
    float*    acc = (float*)   (ws + off); // 4 * BN floats
    float* maxIJ = acc;
    float* sumIJ = acc + BN_;
    float* maxJI = acc + 2 * BN_;
    float* sumJI = acc + 3 * BN_;

    const int T = 256;

    // 0) zero accumulators (must happen every call: graph replay)
    zero_acc_kernel<<<(4 * BN_ + T - 1) / T, T, 0, stream>>>(acc, 4 * BN_);

    // 1) convert weights (transposed) and features to f16
    conv_w_kernel<<<(2 * F_ * H_ + T - 1) / T, T, 0, stream>>>(W1, W2, W1t, W2t);
    {
        size_t nx = (size_t)BN_ * F_;
        conv_x_kernel<<<(unsigned)((nx + T - 1) / T), T, 0, stream>>>(X, Xh, nx);
    }

    // 2) WMMA GEMM (2x2 blocked): Y = [Xh@W1+b1 ; Xh@W2+b2]
    {
        const int tasks = 2 * (BN_ / 32) * (H_ / 32);    // 20000 wave-tasks
        gemm_wmma_kernel<<<tasks / 8, T, 0, stream>>>(Xh, W1t, W2t, b1, b2, Y);
    }

    // 3) in-place LayerNorm -> A1/A2
    ln_kernel<<<(2 * BN_) / 8, T, 0, stream>>>(Y, g1, bb1, g2, bb2);

    // 4-6) edge phases
    edge_pass1_kernel<<<(B_ * E_) / 8, T, 0, stream>>>(ei, Y, W3, b3, W4, b4,
                                                       out, maxIJ, maxJI);
    edge_pass2_kernel<<<(2 * B_ * E_ + T - 1) / T, T, 0, stream>>>(ei, out, maxIJ, maxJI,
                                                                   sumIJ, sumJI);
    edge_pass3_kernel<<<(2 * B_ * E_ + T - 1) / T, T, 0, stream>>>(ei, out, sumIJ, sumJI);
}
